// PatchMatch_78383153152425
// MI455X (gfx1250) — compile-verified, matrix operand order checked by reading, not run on hardware
//
#include <hip/hip_runtime.h>
#include <hip/hip_bf16.h>
#include <math.h>

typedef __attribute__((ext_vector_type(16))) _Float16 v16h;
typedef __attribute__((ext_vector_type(8)))  float    v8f;

#define BB 2
#define GG 8
#define DD 16
#define HH 256
#define WW 320
#define HWW (HH*WW)
#define PIXN (BB*HWW)
#define TPW 16          // pixels per wave tile
#define NWAVE 8         // waves per block (256 threads, wave32)

__device__ __forceinline__ int refl(int i, int n) {
    i = (i < 0) ? -i : i;
    i = (i >= n) ? (2*n - 2 - i) : i;
    return i;
}

// ---------------------------------------------------------------------------
// Kernel 1: PixelwiseNet (max over D -> view weight) + SimilarityNet via WMMA.
// Emits s[pix][d] and xn[pix][d] (d innermost) into workspace.
// ---------------------------------------------------------------------------
__global__ __launch_bounds__(256)
void pm_mlp_kernel(const float* __restrict__ x1,
                   const float* __restrict__ depth_sample,
                   const float* __restrict__ depth_min,
                   const float* __restrict__ depth_max,
                   const float* __restrict__ s_w0, const float* __restrict__ s_g0, const float* __restrict__ s_b0,
                   const float* __restrict__ s_w1, const float* __restrict__ s_g1, const float* __restrict__ s_b1,
                   const float* __restrict__ s_w2, const float* __restrict__ s_bias2,
                   const float* __restrict__ p_w0, const float* __restrict__ p_g0, const float* __restrict__ p_b0,
                   const float* __restrict__ p_w1, const float* __restrict__ p_g1, const float* __restrict__ p_b1,
                   const float* __restrict__ p_w2, const float* __restrict__ p_bias2,
                   float* __restrict__ ws_s, float* __restrict__ ws_xn)
{
    __shared__ _Float16 lds_x[NWAVE][DD][TPW][GG];  // f16 x-tile (A-layout rows), all depths
    __shared__ _Float16 lds_h[NWAVE][16*16];        // hidden-layer transpose staging
    __shared__ float    lds_r[NWAVE][16*8];         // layer-3 partial products

    const int wave = threadIdx.x >> 5;
    const int lane = threadIdx.x & 31;
    const int tilesPerRow = WW / TPW;               // 20
    const int tile = blockIdx.x * NWAVE + wave;     // 10240 tiles total
    const int b  = tile / (HH * tilesPerRow);
    const int r0 = tile % (HH * tilesPerRow);
    const int h  = r0 / tilesPerRow;
    const int w0 = (r0 % tilesPerRow) * TPW;

    const int  n   = lane & 15;                     // N (output channel) for B/C/D layouts
    const bool hiL = lane >= 16;

    // ---- B-matrices (weights) in the 16-bit 32x16 WMMA layout ----
    // lanes 0-15 hold K=0..7 (halves 0..7) / K=16..23 (halves 8..15, zero-padded);
    // lanes 16-31 hold K=8..15 / K=24..31.
    v16h b1s = {}, b1p = {}, b2s = {}, b2p = {};
    if (!hiL) {
#pragma unroll
        for (int j = 0; j < 8; ++j) {               // layer 1: K = input channel < 8
            b1s[j] = (_Float16)s_w0[n*8 + j];
            b1p[j] = (_Float16)p_w0[n*8 + j];
        }
    }
    if (n < 8) {
        const int kb = hiL ? 8 : 0;                 // layer 2: K = hidden channel < 16
#pragma unroll
        for (int j = 0; j < 8; ++j) {
            b2s[j] = (_Float16)s_w1[n*16 + kb + j];
            b2p[j] = (_Float16)p_w1[n*16 + kb + j];
        }
    }

    const float rs   = rsqrtf(1.0f + 1e-5f);        // BN at init stats
    const float g0s_ = s_g0[n]*rs, c0s_ = s_b0[n];
    const float g0p_ = p_g0[n]*rs, c0p_ = p_b0[n];
    const float g1s_ = (n < 8 ? s_g1[n] : 0.f)*rs, c1s_ = (n < 8 ? s_b1[n] : 0.f);
    const float g1p_ = (n < 8 ? p_g1[n] : 0.f)*rs, c1p_ = (n < 8 ? p_b1[n] : 0.f);
    const float w2s_ = (n < 8 ? s_w2[n] : 0.f);
    const float w2p_ = (n < 8 ? p_w2[n] : 0.f);
    const float bias2s = s_bias2[0], bias2p = p_bias2[0];

    const float invmin   = 1.0f / depth_min[b];
    const float invmax   = 1.0f / depth_max[b];
    const float invrange = 1.0f / (invmin - invmax);

    float vw = -1e30f;                              // view-weight running max (lanes 0-15)

    // ====================== pass 1: PixelwiseNet ======================
    for (int d = 0; d < DD; ++d) {
        v16h a = {};
        if (!hiL) {                                 // A: M=lane position, K=channel
            const int base = (b*GG*DD + d)*HWW + h*WW + w0 + lane;
#pragma unroll
            for (int c = 0; c < GG; ++c) {
                _Float16 xh = (_Float16)x1[base + c*DD*HWW];
                a[c] = xh;
                lds_x[wave][d][lane][c] = xh;       // cache f16 tile for pass 2
            }
        }
        v8f acc = {};
        acc = __builtin_amdgcn_wmma_f32_16x16x32_f16(false, a, false, b1p,
                                                     (short)0, acc, false, false);
#pragma unroll
        for (int r = 0; r < 8; ++r) {               // D: row M=r+8*hiL, col N=n
            float v = fmaxf(g0p_*acc[r] + c0p_, 0.f);
            lds_h[wave][(r + (hiL ? 8 : 0))*16 + n] = (_Float16)v;
        }
        __syncthreads();
        v16h a2 = {};
        {
            const _Float16* row = &lds_h[wave][(lane & 15)*16 + (hiL ? 8 : 0)];
#pragma unroll
            for (int j = 0; j < 8; ++j) a2[j] = row[j];
        }
        v8f acc2 = {};
        acc2 = __builtin_amdgcn_wmma_f32_16x16x32_f16(false, a2, false, b2p,
                                                      (short)0, acc2, false, false);
        if (n < 8) {
#pragma unroll
            for (int r = 0; r < 8; ++r) {
                float v = fmaxf(g1p_*acc2[r] + c1p_, 0.f);
                lds_r[wave][(r + (hiL ? 8 : 0))*8 + n] = v * w2p_;
            }
        }
        __syncthreads();
        if (!hiL) {                                 // layer 3 + sigmoid + max over D
            float o = bias2p;
#pragma unroll
            for (int c = 0; c < 8; ++c) o += lds_r[wave][lane*8 + c];
            float sg = 1.0f / (1.0f + __expf(-o));
            vw = fmaxf(vw, sg);
        }
    }

    __syncthreads();

    // ====================== pass 2: SimilarityNet =====================
    for (int d = 0; d < DD; ++d) {
        v16h a = {};
        if (!hiL) {
#pragma unroll
            for (int c = 0; c < GG; ++c)
                a[c] = (_Float16)((float)lds_x[wave][d][lane][c] * vw);
        }
        v8f acc = {};
        acc = __builtin_amdgcn_wmma_f32_16x16x32_f16(false, a, false, b1s,
                                                     (short)0, acc, false, false);
#pragma unroll
        for (int r = 0; r < 8; ++r) {
            float v = fmaxf(g0s_*acc[r] + c0s_, 0.f);
            lds_h[wave][(r + (hiL ? 8 : 0))*16 + n] = (_Float16)v;
        }
        __syncthreads();
        v16h a2 = {};
        {
            const _Float16* row = &lds_h[wave][(lane & 15)*16 + (hiL ? 8 : 0)];
#pragma unroll
            for (int j = 0; j < 8; ++j) a2[j] = row[j];
        }
        v8f acc2 = {};
        acc2 = __builtin_amdgcn_wmma_f32_16x16x32_f16(false, a2, false, b2s,
                                                      (short)0, acc2, false, false);
        if (n < 8) {
#pragma unroll
            for (int r = 0; r < 8; ++r) {
                float v = fmaxf(g1s_*acc2[r] + c1s_, 0.f);
                lds_r[wave][(r + (hiL ? 8 : 0))*8 + n] = v * w2s_;
            }
        }
        __syncthreads();
        if (!hiL) {
            float o = bias2s;
#pragma unroll
            for (int c = 0; c < 8; ++c) o += lds_r[wave][lane*8 + c];
            const int pix = b*HWW + h*WW + w0 + lane;
            ws_s[pix*DD + d] = o;
            float dsv = depth_sample[(b*DD + d)*HWW + h*WW + w0 + lane];
            ws_xn[pix*DD + d] = (1.0f/dsv - invmax) * invrange;
        }
    }
}

// ---------------------------------------------------------------------------
// Kernel 2: 9-neighbor two-dilation gather (reflect pad), depth weight,
// softmax over D, depth regression. One thread per output pixel.
// ---------------------------------------------------------------------------
__global__ __launch_bounds__(256)
void pm_gather_kernel(const float* __restrict__ offset,
                      const float* __restrict__ depth_sample,
                      const float* __restrict__ ws_s,
                      const float* __restrict__ ws_xn,
                      float* __restrict__ out)
{
    const int pix = blockIdx.x * blockDim.x + threadIdx.x;
    const int b  = pix / HWW;
    const int hw = pix % HWW;
    const int h  = hw / WW;
    const int w  = hw % WW;

    float off[18];
#pragma unroll
    for (int s = 0; s < 18; ++s) off[s] = offset[(b*18 + s)*HWW + hw];

    float xg[DD], sg[DD], xnc[DD];
#pragma unroll
    for (int d = 0; d < DD; ++d) { xg[d] = 0.f; sg[d] = 0.f; }
    {
        const float* p = ws_xn + (size_t)pix*DD;
#pragma unroll
        for (int d = 0; d < DD; ++d) xnc[d] = p[d];
    }

    for (int s9 = 0; s9 < 9; ++s9) {
        const int dy = (s9/3)*2 - 2;                // {-2,0,2}
        const int dx = (s9%3)*2 - 2;
        const int h1 = refl(h +   dy, HH), w1 = refl(w +   dx, WW);
        const int h2 = refl(h + 2*dy, HH), w2 = refl(w + 2*dx, WW);  // {-4,0,4}
        const float* xn1 = ws_xn + (size_t)(b*HWW + h1*WW + w1)*DD;
        const float* xn2 = ws_xn + (size_t)(b*HWW + h2*WW + w2)*DD;
        const float* sb1 = ws_s  + (size_t)(b*HWW + h1*WW + w1)*DD;
        const float* sb2 = ws_s  + (size_t)(b*HWW + h2*WW + w2)*DD;
        const float o1 = 0.5f * off[s9 + 9];        // dilation-1 weight
        const float o2 = 0.5f * off[s9];            // dilation-2 weight
#pragma unroll
        for (int d = 0; d < DD; ++d) {
            xg[d] += o1*xn1[d] + o2*xn2[d];
            sg[d] += o1*sb1[d] + o2*sb2[d];
        }
    }

    float score[DD];
    float m = -1e30f;
#pragma unroll
    for (int d = 0; d < DD; ++d) {
        float t = fabsf(xg[d] - xnc[d]) * 40.0f;    // / 0.025
        t = fminf(t, 4.0f);
        float dw = 1.0f / (1.0f + __expf(-(2.0f - t)*2.0f));
        score[d] = sg[d] * dw;
        m = fmaxf(m, score[d]);
    }
    float sum = 0.f, num = 0.f;
#pragma unroll
    for (int d = 0; d < DD; ++d) {
        float e = __expf(score[d] - m);
        sum += e;
        num += e * depth_sample[(b*DD + d)*HWW + hw];
    }
    out[pix] = num / sum;
}

// ---------------------------------------------------------------------------
extern "C" void kernel_launch(void* const* d_in, const int* in_sizes, int n_in,
                              void* d_out, int out_size, void* d_ws, size_t ws_size,
                              hipStream_t stream) {
    const float* x1           = (const float*)d_in[0];
    const float* offset       = (const float*)d_in[1];
    const float* depth_sample = (const float*)d_in[2];
    const float* depth_min    = (const float*)d_in[3];
    const float* depth_max    = (const float*)d_in[4];
    const float* s_w0    = (const float*)d_in[5];
    const float* s_g0    = (const float*)d_in[6];
    const float* s_b0    = (const float*)d_in[7];
    const float* s_w1    = (const float*)d_in[8];
    const float* s_g1    = (const float*)d_in[9];
    const float* s_b1    = (const float*)d_in[10];
    const float* s_w2    = (const float*)d_in[11];
    const float* s_bias2 = (const float*)d_in[12];
    const float* p_w0    = (const float*)d_in[13];
    const float* p_g0    = (const float*)d_in[14];
    const float* p_b0    = (const float*)d_in[15];
    const float* p_w1    = (const float*)d_in[16];
    const float* p_g1    = (const float*)d_in[17];
    const float* p_b1    = (const float*)d_in[18];
    const float* p_w2    = (const float*)d_in[19];
    const float* p_bias2 = (const float*)d_in[20];

    float* ws_s  = (float*)d_ws;                    // [PIX][D]
    float* ws_xn = ws_s + (size_t)PIXN*DD;          // [PIX][D]  (~21 MB total)

    const int tiles  = BB * HH * (WW / TPW);        // 10240 waves
    const int blocks = tiles / NWAVE;               // 1280
    pm_mlp_kernel<<<blocks, 256, 0, stream>>>(
        x1, depth_sample, depth_min, depth_max,
        s_w0, s_g0, s_b0, s_w1, s_g1, s_b1, s_w2, s_bias2,
        p_w0, p_g0, p_b0, p_w1, p_g1, p_b1, p_w2, p_bias2,
        ws_s, ws_xn);

    pm_gather_kernel<<<PIXN/256, 256, 0, stream>>>(
        offset, depth_sample, ws_s, ws_xn, (float*)d_out);
}
